// PlanStack_16793322127884
// MI455X (gfx1250) — matrix operand order repaired and checked by reading, not compiled, not run on hardware
//
#include <hip/hip_runtime.h>

typedef __attribute__((ext_vector_type(8)))  __bf16 v8bf;
typedef __attribute__((ext_vector_type(16))) __bf16 v16bf;
typedef __attribute__((ext_vector_type(8)))  float  v8f;

#define BB 16384
#define DD 8
#define HH 1024
#define EPSF 1e-5f

// ---------------------------------------------------------------------------
// Kernel 0: split f32 -> (bf16 hi, bf16 lo) with lo = bf16(x - f32(hi)).
// ---------------------------------------------------------------------------
__global__ __launch_bounds__(256) void split_bf16(
    const float* __restrict__ x, __bf16* __restrict__ hi,
    __bf16* __restrict__ lo, int n)
{
    int i = blockIdx.x * blockDim.x + threadIdx.x;
    if (i >= n) return;
    float v = x[i];
    __bf16 h = (__bf16)v;                 // RNE convert
    hi[i] = h;
    lo[i] = (__bf16)(v - (float)h);       // residual
}

__device__ __forceinline__ v16bf cat16(v8bf a, v8bf b) {
    return __builtin_shufflevector(a, b, 0,1,2,3,4,5,6,7,8,9,10,11,12,13,14,15);
}

// ---------------------------------------------------------------------------
// Kernel 1: push_raw = hidden @ W_push^T + b_push  via bf16x3 emulation on
// v_wmma_f32_16x16x32_bf16 (f32 accumulate):
//   A*B ~= Ahi*Bhi + Ahi*Blo + Alo*Bhi
// grid (H/64, B/128), block 256 (8 waves). Wave w: rows m0=by*128+w*16,
// cols n0=bx*64 (4 N-tiles). K loop: 32 chunks of K=32, 12 wmma each.
// A frag (16x32 bf16): lane l -> M=l&15; elems 0-15: K=k+16*(l>>4)+0..15
//                      (two b128 loads)
// B frag (32x16 bf16): lane l -> N=l&15; elems 0-15: K=k+16*(l>>4)+0..15
//                      (two b128 loads; B[k][n]=W[n][k], row-major W)
// C/D (16x16 f32): vgpr r, lane l -> M=r+8*(l>>4), N=l&15
// ---------------------------------------------------------------------------
__global__ __launch_bounds__(256) void push_gemm_bf16x3(
    const __bf16* __restrict__ Ahi, const __bf16* __restrict__ Alo,
    const __bf16* __restrict__ Whi, const __bf16* __restrict__ Wlo,
    const float* __restrict__ bp, float* __restrict__ out)
{
    const int wave = threadIdx.x >> 5;
    const int lane = threadIdx.x & 31;
    const int m0   = blockIdx.y * 128 + wave * 16;
    const int n0   = blockIdx.x * 64;
    const int nl   = lane & 15;
    const int half = lane >> 4;

    const int aoff0 = half * 16;       // A elems 0-7 K-offset
    const int aoff1 = half * 16 + 8;   // A elems 8-15 K-offset
    const int boff  = half * 16;       // B elems 0-15 K-offset

    const __bf16* Ah = Ahi + (size_t)(m0 + nl) * HH;
    const __bf16* Al = Alo + (size_t)(m0 + nl) * HH;
    const __bf16* Wh0 = Whi + (size_t)(n0 +  0 + nl) * HH;
    const __bf16* Wh1 = Whi + (size_t)(n0 + 16 + nl) * HH;
    const __bf16* Wh2 = Whi + (size_t)(n0 + 32 + nl) * HH;
    const __bf16* Wh3 = Whi + (size_t)(n0 + 48 + nl) * HH;
    const __bf16* Wl0 = Wlo + (size_t)(n0 +  0 + nl) * HH;
    const __bf16* Wl1 = Wlo + (size_t)(n0 + 16 + nl) * HH;
    const __bf16* Wl2 = Wlo + (size_t)(n0 + 32 + nl) * HH;
    const __bf16* Wl3 = Wlo + (size_t)(n0 + 48 + nl) * HH;

    v8f acc0 = {}, acc1 = {}, acc2 = {}, acc3 = {};

    #pragma unroll 2
    for (int k = 0; k < HH; k += 32) {
        v16bf ah = cat16(*(const v8bf*)(Ah + k + aoff0), *(const v8bf*)(Ah + k + aoff1));
        v16bf al = cat16(*(const v8bf*)(Al + k + aoff0), *(const v8bf*)(Al + k + aoff1));

        v16bf bh0 = cat16(*(const v8bf*)(Wh0 + k + boff), *(const v8bf*)(Wh0 + k + boff + 8));
        v16bf bl0 = cat16(*(const v8bf*)(Wl0 + k + boff), *(const v8bf*)(Wl0 + k + boff + 8));
        acc0 = __builtin_amdgcn_wmma_f32_16x16x32_bf16(false, ah, false, bh0, (short)0, acc0, false, false);
        acc0 = __builtin_amdgcn_wmma_f32_16x16x32_bf16(false, ah, false, bl0, (short)0, acc0, false, false);
        acc0 = __builtin_amdgcn_wmma_f32_16x16x32_bf16(false, al, false, bh0, (short)0, acc0, false, false);

        v16bf bh1 = cat16(*(const v8bf*)(Wh1 + k + boff), *(const v8bf*)(Wh1 + k + boff + 8));
        v16bf bl1 = cat16(*(const v8bf*)(Wl1 + k + boff), *(const v8bf*)(Wl1 + k + boff + 8));
        acc1 = __builtin_amdgcn_wmma_f32_16x16x32_bf16(false, ah, false, bh1, (short)0, acc1, false, false);
        acc1 = __builtin_amdgcn_wmma_f32_16x16x32_bf16(false, ah, false, bl1, (short)0, acc1, false, false);
        acc1 = __builtin_amdgcn_wmma_f32_16x16x32_bf16(false, al, false, bh1, (short)0, acc1, false, false);

        v16bf bh2 = cat16(*(const v8bf*)(Wh2 + k + boff), *(const v8bf*)(Wh2 + k + boff + 8));
        v16bf bl2 = cat16(*(const v8bf*)(Wl2 + k + boff), *(const v8bf*)(Wl2 + k + boff + 8));
        acc2 = __builtin_amdgcn_wmma_f32_16x16x32_bf16(false, ah, false, bh2, (short)0, acc2, false, false);
        acc2 = __builtin_amdgcn_wmma_f32_16x16x32_bf16(false, ah, false, bl2, (short)0, acc2, false, false);
        acc2 = __builtin_amdgcn_wmma_f32_16x16x32_bf16(false, al, false, bh2, (short)0, acc2, false, false);

        v16bf bh3 = cat16(*(const v8bf*)(Wh3 + k + boff), *(const v8bf*)(Wh3 + k + boff + 8));
        v16bf bl3 = cat16(*(const v8bf*)(Wl3 + k + boff), *(const v8bf*)(Wl3 + k + boff + 8));
        acc3 = __builtin_amdgcn_wmma_f32_16x16x32_bf16(false, ah, false, bh3, (short)0, acc3, false, false);
        acc3 = __builtin_amdgcn_wmma_f32_16x16x32_bf16(false, ah, false, bl3, (short)0, acc3, false, false);
        acc3 = __builtin_amdgcn_wmma_f32_16x16x32_bf16(false, al, false, bh3, (short)0, acc3, false, false);
    }

    // Epilogue: add bias, store. vgpr r -> row m0 + r + 8*half, col n.
    const int mh = half * 8;
    {
        const int col = n0 + 0 * 16 + nl; const float bias = bp[col];
        #pragma unroll
        for (int r = 0; r < 8; ++r) out[(size_t)(m0 + mh + r) * HH + col] = acc0[r] + bias;
    }
    {
        const int col = n0 + 1 * 16 + nl; const float bias = bp[col];
        #pragma unroll
        for (int r = 0; r < 8; ++r) out[(size_t)(m0 + mh + r) * HH + col] = acc1[r] + bias;
    }
    {
        const int col = n0 + 2 * 16 + nl; const float bias = bp[col];
        #pragma unroll
        for (int r = 0; r < 8; ++r) out[(size_t)(m0 + mh + r) * HH + col] = acc2[r] + bias;
    }
    {
        const int col = n0 + 3 * 16 + nl; const float bias = bp[col];
        #pragma unroll
        for (int r = 0; r < 8; ++r) out[(size_t)(m0 + mh + r) * HH + col] = acc3[r] + bias;
    }
}

// ---------------------------------------------------------------------------
// Kernel 2: one block (256 threads) per batch row; float4 (b128) streaming.
// Each thread owns columns [4*tid, 4*tid+4), so the in-place read->rewrite of
// the 'top' region (push_raw -> final top) is race-free.
// ---------------------------------------------------------------------------
__global__ __launch_bounds__(256) void stack_update(
    const float* __restrict__ hidden, const float* __restrict__ stack,
    const float* __restrict__ pointer, const float* __restrict__ ln_g,
    const float* __restrict__ ln_b, const float* __restrict__ w_pop,
    const float* __restrict__ b_pop,
    float* __restrict__ new_stack, float* __restrict__ new_pointer,
    float* __restrict__ top, float* __restrict__ popped)
{
    const int b   = blockIdx.x;
    const int tid = threadIdx.x;

    __shared__ float red1[256];
    __shared__ float red2[256];
    __shared__ float sh_logit;

    // --- pop gate dot product ---
    const float4 hv = ((const float4*)(hidden + (size_t)b * HH))[tid];
    const float4 wv = ((const float4*)w_pop)[tid];
    red1[tid] = hv.x * wv.x + hv.y * wv.y + hv.z * wv.z + hv.w * wv.w;
    __syncthreads();
    #pragma unroll
    for (int s = 128; s > 0; s >>= 1) {
        if (tid < s) red1[tid] += red1[tid + s];
        __syncthreads();
    }
    if (tid == 0) sh_logit = red1[0] + b_pop[0];
    __syncthreads();
    const bool is_pop = sh_logit > 0.f;   // sigmoid(x) > 0.5  <=>  x > 0

    // --- LayerNorm of push_raw (in 'top' region from kernel 1) ---
    const float4 xv = ((const float4*)(top + (size_t)b * HH))[tid];
    red1[tid] = xv.x + xv.y + xv.z + xv.w;
    red2[tid] = xv.x * xv.x + xv.y * xv.y + xv.z * xv.z + xv.w * xv.w;
    __syncthreads();
    #pragma unroll
    for (int s = 128; s > 0; s >>= 1) {
        if (tid < s) { red1[tid] += red1[tid + s]; red2[tid] += red2[tid + s]; }
        __syncthreads();
    }
    const float mu  = red1[0] * (1.0f / HH);
    const float var = red2[0] * (1.0f / HH) - mu * mu;
    const float rs  = rsqrtf(var + EPSF);

    const float4 g4 = ((const float4*)ln_g)[tid];
    const float4 o4 = ((const float4*)ln_b)[tid];
    float4 p;
    p.x = (xv.x - mu) * rs * g4.x + o4.x;
    p.y = (xv.y - mu) * rs * g4.y + o4.y;
    p.z = (xv.z - mu) * rs * g4.z + o4.z;
    p.w = (xv.w - mu) * rs * g4.w + o4.w;

    // --- stack pointer logic ---
    const int ptr      = (int)pointer[b];
    const bool do_pop  = is_pop && (ptr > 0);
    const bool do_push = (!is_pop) && (ptr < DD);
    const int prevIdx  = (ptr - 1) > 0 ? (ptr - 1) : 0;

    const float4 pv = ((const float4*)(stack + ((size_t)b * DD + prevIdx) * HH))[tid];
    const float4 z4 = make_float4(0.f, 0.f, 0.f, 0.f);

    // --- top & popped ---
    float4 t;
    if (do_pop || (!do_push && ptr > 0)) t = pv;
    else if (do_push)                    t = p;
    else                                 t = z4;
    ((float4*)(top    + (size_t)b * HH))[tid] = t;
    ((float4*)(popped + (size_t)b * HH))[tid] = do_pop ? pv : z4;

    // --- new_stack: copy + masked scatter of push row at slot ptr ---
    const float* sbase = stack     + (size_t)b * DD * HH;
    float*       nbase = new_stack + (size_t)b * DD * HH;
    #pragma unroll
    for (int d = 0; d < DD; ++d) {
        float4 v = ((const float4*)(sbase + (size_t)d * HH))[tid];
        if (do_push && d == ptr) v = p;
        ((float4*)(nbase + (size_t)d * HH))[tid] = v;
    }

    // --- new_pointer ---
    if (tid == 0) {
        int np = do_pop ? (ptr - 1) : (do_push ? (ptr + 1) : ptr);
        new_pointer[b] = (float)np;
    }
}

extern "C" void kernel_launch(void* const* d_in, const int* in_sizes, int n_in,
                              void* d_out, int out_size, void* d_ws, size_t ws_size,
                              hipStream_t stream) {
    (void)in_sizes; (void)n_in; (void)out_size; (void)d_ws; (void)ws_size;

    const float* hidden  = (const float*)d_in[0];
    const float* stack   = (const float*)d_in[1];
    const float* pointer = (const float*)d_in[2];
    const float* W_push  = (const float*)d_in[3];
    const float* b_push  = (const float*)d_in[4];
    const float* ln_g    = (const float*)d_in[5];
    const float* ln_b    = (const float*)d_in[6];
    const float* w_pop   = (const float*)d_in[7];
    const float* b_pop   = (const float*)d_in[8];

    float* out         = (float*)d_out;
    float* new_stack   = out;                                  // B*D*H
    float* new_pointer = out + (size_t)BB * DD * HH;           // B
    float* top         = new_pointer + BB;                     // B*H
    float* popped      = top + (size_t)BB * HH;                // B*H

    // bf16 scratch lives inside output regions that kernel 2 rewrites last:
    //   hidden hi/lo (2 x 32 MB) exactly fills the 'popped' region (64 MB)
    //   W hi/lo (2 x 2 MB) at the head of 'new_stack' (512 MB)
    __bf16* hidden_hi = (__bf16*)popped;
    __bf16* hidden_lo = hidden_hi + (size_t)BB * HH;
    __bf16* W_hi      = (__bf16*)new_stack;
    __bf16* W_lo      = W_hi + (size_t)HH * HH;

    split_bf16<<<(BB * HH) / 256, 256, 0, stream>>>(hidden, hidden_hi, hidden_lo, BB * HH);
    split_bf16<<<(HH * HH) / 256, 256, 0, stream>>>(W_push, W_hi, W_lo, HH * HH);

    // Kernel 1: push_raw -> 'top' region (scratch; rewritten by kernel 2)
    dim3 g1(HH / 64, BB / 128);
    push_gemm_bf16x3<<<g1, 256, 0, stream>>>(hidden_hi, hidden_lo, W_hi, W_lo, b_push, top);

    // Kernel 2: everything else, one block per row
    stack_update<<<BB, 256, 0, stream>>>(hidden, stack, pointer, ln_g, ln_b,
                                         w_pop, b_pop,
                                         new_stack, new_pointer, top, popped);
}
